// Decoder_76381698392814
// MI455X (gfx1250) — matrix operand order repaired and checked by reading, not compiled, not run on hardware
//
#include <hip/hip_runtime.h>
#include <hip/hip_bf16.h>
#include <stdint.h>

// ---------------- problem dims ----------------
#define B_      64
#define T_      32
#define S_      64
#define V_      32000
#define E_      512
#define U_      1024
#define K_LSTM  2560            // U + E + U
#define N_LSTM  4096            // 4*U
#define KT_LSTM 80              // 2560/32
#define NT_LSTM 256             // 4096/16
#define KT_OUT  32              // 1024/32
#define NT_OUT  2000            // 32000/16

typedef __attribute__((ext_vector_type(16))) __bf16 v16bf;
typedef __attribute__((ext_vector_type(8)))  float  v8f;

union FragBf { v16bf v; uint4 u[2]; };

__device__ __forceinline__ void load_frag(FragBf& f, const unsigned short* p, int d) {
  f.u[0] = *(const uint4*)(p);
  f.u[1] = *(const uint4*)(p + d);
}

__device__ __forceinline__ unsigned short f2bf(float f) {
  unsigned int u = __float_as_uint(f);
  u += 0x7FFFu + ((u >> 16) & 1u);            // round-to-nearest-even
  return (unsigned short)(u >> 16);
}

__device__ __forceinline__ float sigmoidf_(float x) {
  return 1.0f / (1.0f + __expf(-x));
}
__device__ __forceinline__ float tanhf_(float x) {
  return 1.0f - 2.0f / (__expf(2.0f * x) + 1.0f);
}

__device__ __forceinline__ v8f wmma_bf16(const FragBf& a, const FragBf& b, v8f c) {
  return __builtin_amdgcn_wmma_f32_16x16x32_bf16(false, a.v, false, b.v, (short)0, c,
                                                 false, false);
}

// -------- one-time: convert [Wk;Wr] -> bf16, pre-swizzled B-fragment layout --------
// dst[((kT*NT + nT)*32 + lane)*16 + half] = W[kT*32 + kk][nT*16 + nn]
//   kk = half + ((half>=8) ? 8 : 0) + ((lane<16) ? 0 : 8) ; nn = lane&15
__global__ void k_swizzle_wkr(const float* __restrict__ Wk, const float* __restrict__ Wr,
                              unsigned short* __restrict__ dst) {
  long long idx = (long long)blockIdx.x * blockDim.x + threadIdx.x; // < 2560*4096
  int hh   = (int)(idx & 15);
  int lane = (int)((idx >> 4) & 31);
  long long rest = idx >> 9;
  int nT = (int)(rest % NT_LSTM);
  int kT = (int)(rest / NT_LSTM);
  int k = kT * 32 + hh + ((hh >= 8) ? 8 : 0) + ((lane < 16) ? 0 : 8);
  int n = nT * 16 + (lane & 15);
  float v = (k < (U_ + E_)) ? Wk[(long long)k * N_LSTM + n]
                            : Wr[(long long)(k - (U_ + E_)) * N_LSTM + n];
  dst[idx] = f2bf(v);
}

__global__ void k_swizzle_wd(const float* __restrict__ Wd, unsigned short* __restrict__ dst) {
  long long idx = (long long)blockIdx.x * blockDim.x + threadIdx.x; // < 1024*32000
  int hh   = (int)(idx & 15);
  int lane = (int)((idx >> 4) & 31);
  long long rest = idx >> 9;
  int nT = (int)(rest % NT_OUT);
  int kT = (int)(rest / NT_OUT);
  int k = kT * 32 + hh + ((hh >= 8) ? 8 : 0) + ((lane < 16) ? 0 : 8);
  int n = nT * 16 + (lane & 15);
  dst[idx] = f2bf(Wd[(long long)k * V_ + n]);
}

__global__ void k_init(const float* __restrict__ h0, const float* __restrict__ c0,
                       float* __restrict__ h, float* __restrict__ c) {
  int i = blockIdx.x * blockDim.x + threadIdx.x;   // 64*1024
  h[i] = h0[i];
  c[i] = c0[i];
}

// -------- per step: attention + build xcat = [ctx | emb[tok] | h] as bf16 --------
__global__ void k_attn(const float* __restrict__ enc, const float* __restrict__ h_state,
                       const int* __restrict__ tokens, const float* __restrict__ emb,
                       unsigned short* __restrict__ xcat, int t) {
  __shared__ float sh_h[U_];
  __shared__ float sh_sc[S_];
  __shared__ float sh_red[2];
  const int b   = blockIdx.x;
  const int tid = threadIdx.x;                 // 256
  const float* encb = enc + (long long)b * S_ * U_;

  for (int u = tid; u < U_; u += 256) sh_h[u] = h_state[b * U_ + u];
  __syncthreads();

  // scores: wave wv handles rows wv*8..wv*8+7, lanes stride columns (coalesced)
  const int wv = tid >> 5, ln = tid & 31;
  for (int r = 0; r < 8; ++r) {
    const int s = wv * 8 + r;
    const float* row = encb + s * U_;
    float acc = 0.f;
    for (int i = ln; i < U_; i += 32) acc += sh_h[i] * row[i];
    for (int m = 16; m >= 1; m >>= 1) acc += __shfl_xor(acc, m, 32);
    if (ln == 0) sh_sc[s] = acc;
  }
  __syncthreads();
  if (tid == 0) {                              // softmax max
    float mx = sh_sc[0];
    for (int s = 1; s < S_; ++s) mx = fmaxf(mx, sh_sc[s]);
    sh_red[0] = mx;
  }
  __syncthreads();
  if (tid < S_) sh_sc[tid] = __expf(sh_sc[tid] - sh_red[0]);
  __syncthreads();
  if (tid == 0) {                              // softmax sum
    float sm = 0.f;
    for (int s = 0; s < S_; ++s) sm += sh_sc[s];
    sh_red[1] = 1.0f / sm;
  }
  __syncthreads();
  if (tid < S_) sh_sc[tid] *= sh_red[1];
  __syncthreads();

  // ctx (coalesced over u) + concat writes
  unsigned short* xb = xcat + (long long)b * K_LSTM;
  for (int u = tid; u < U_; u += 256) {
    float acc = 0.f;
    for (int s = 0; s < S_; ++s) acc += sh_sc[s] * encb[s * U_ + u];
    xb[u] = f2bf(acc);
  }
  const int tok = tokens[b * T_ + t];
  const float* erow = emb + (long long)tok * E_;
  for (int e = tid; e < E_; e += 256) xb[U_ + e] = f2bf(erow[e]);
  for (int u = tid; u < U_; u += 256) xb[U_ + E_ + u] = f2bf(sh_h[u]);
}

// -------- per step: z = xcat @ [Wk;Wr] + b, fused LSTM gates -> h,c --------
// grid 64 (16 gate-columns each), block 256 = 8 waves:
//   wave = (mTile = wv&3, kHalf = wv>>2); kHalf 1 dumps partials to LDS,
//   kHalf 0 reduces and does fused gate math.
// K loop: explicit ping-pong pipeline (2 k-steps/iter, no register copies).
__global__ void k_lstm(const unsigned short* __restrict__ xcat,
                       const unsigned short* __restrict__ Wkr,
                       const float* __restrict__ bl,
                       float* __restrict__ h_state, float* __restrict__ c_state,
                       unsigned short* __restrict__ h_bf) {
  __shared__ float sh[4 * 4 * 8 * 32];           // 16 KB: [mT][gate][r][lane]
  const int bx  = blockIdx.x;                    // column group 0..63
  const int tid = threadIdx.x;
  const int wv  = tid >> 5;
  const int ln  = tid & 31;
  const int mT  = wv & 3;
  const int kh  = wv >> 2;                       // K half: 0 or 1
  const int m0  = mT * 16;
  const int mrow = m0 + (ln & 15);
  const int kofs = (ln >> 4) * 8;
  const int ncol = bx * 16 + (ln & 15);
  const int kt0  = kh * (KT_LSTM / 2);           // 0 or 40
  const int KH   = KT_LSTM / 2;                  // 40 (even)

  v8f acc[4];
  #pragma unroll
  for (int g = 0; g < 4; ++g) {
    const float bias = (kh == 0) ? bl[g * U_ + ncol] : 0.0f;
    #pragma unroll
    for (int i = 0; i < 8; ++i) acc[g][i] = bias;
  }

  const unsigned short* aBase = xcat + (long long)mrow * K_LSTM + kofs;
  const unsigned short* bBase = Wkr + (long long)ln * 16;

  FragBf A0, A1, B0[4], B1[4];
  load_frag(A0, aBase + kt0 * 32, 16);
  #pragma unroll
  for (int g = 0; g < 4; ++g)
    load_frag(B0[g], bBase + (long long)(kt0 * NT_LSTM + g * 64 + bx) * 512, 8);

  for (int i = 0; i < KH; i += 2) {
    const int k1 = kt0 + i + 1;
    load_frag(A1, aBase + k1 * 32, 16);          // issue buf1 loads
    #pragma unroll
    for (int g = 0; g < 4; ++g)
      load_frag(B1[g], bBase + (long long)(k1 * NT_LSTM + g * 64 + bx) * 512, 8);
    #pragma unroll
    for (int g = 0; g < 4; ++g)                  // consume buf0
      acc[g] = wmma_bf16(A0, B0[g], acc[g]);
    if (i + 2 < KH) {
      const int k2 = kt0 + i + 2;
      load_frag(A0, aBase + k2 * 32, 16);        // issue buf0 loads
      #pragma unroll
      for (int g = 0; g < 4; ++g)
        load_frag(B0[g], bBase + (long long)(k2 * NT_LSTM + g * 64 + bx) * 512, 8);
    }
    #pragma unroll
    for (int g = 0; g < 4; ++g)                  // consume buf1
      acc[g] = wmma_bf16(A1, B1[g], acc[g]);
  }

  if (kh == 1) {                                 // dump partial sums
    #pragma unroll
    for (int g = 0; g < 4; ++g)
      #pragma unroll
      for (int r = 0; r < 8; ++r)
        sh[((mT * 4 + g) * 8 + r) * 32 + ln] = acc[g][r];
  }
  __syncthreads();
  if (kh == 0) {
    #pragma unroll
    for (int g = 0; g < 4; ++g)
      #pragma unroll
      for (int r = 0; r < 8; ++r)
        acc[g][r] += sh[((mT * 4 + g) * 8 + r) * 32 + ln];

    // fused gate math; C layout: M = r + 8*(lane>=16), N = lane&15
    #pragma unroll
    for (int r = 0; r < 8; ++r) {
      const int brow = m0 + r + ((ln >= 16) ? 8 : 0);
      const long long idx = (long long)brow * U_ + ncol;
      const float si = sigmoidf_(acc[0][r]);
      const float sf = sigmoidf_(acc[1][r]);
      const float tg = tanhf_(acc[2][r]);
      const float so = sigmoidf_(acc[3][r]);
      const float cn = sf * c_state[idx] + si * tg;
      const float hn = so * tanhf_(cn);
      c_state[idx] = cn;
      h_state[idx] = hn;
      h_bf[idx]    = f2bf(hn);
    }
  }
}

// -------- per step: out = h_new @ Wd + bd --------
// grid 250, block 256 (8 waves): each wave owns one N-tile, iterates all 4 M-tiles
// (B fragment reused across 4 WMMAs); explicit ping-pong pipeline, Wd prefetch.
__global__ void k_out(const unsigned short* __restrict__ h_bf,
                      const unsigned short* __restrict__ Wd,
                      const float* __restrict__ bd,
                      float* __restrict__ out, int t) {
  const int tid = threadIdx.x;
  const int wv  = tid >> 5;
  const int ln  = tid & 31;
  const int nT  = blockIdx.x * 8 + wv;           // 0..1999
  const int kofs = (ln >> 4) * 8;
  const int ncol = nT * 16 + (ln & 15);

  v8f acc[4];
  const float bias = bd[ncol];
  #pragma unroll
  for (int m = 0; m < 4; ++m)
    #pragma unroll
    for (int i = 0; i < 8; ++i) acc[m][i] = bias;

  const unsigned short* aBase = h_bf + (long long)(ln & 15) * U_ + kofs;
  const unsigned short* bBase = Wd + ((long long)nT * 32 + ln) * 16;

  FragBf A0[4], A1[4], Bb0, Bb1;
  load_frag(Bb0, bBase, 8);
  #pragma unroll
  for (int m = 0; m < 4; ++m) load_frag(A0[m], aBase + m * 16 * U_, 16);

  for (int kt = 0; kt < KT_OUT; kt += 2) {
    const int k1 = kt + 1;
    load_frag(Bb1, bBase + (long long)k1 * NT_OUT * 512, 8);   // issue buf1 loads
    #pragma unroll
    for (int m = 0; m < 4; ++m)
      load_frag(A1[m], aBase + m * 16 * U_ + k1 * 32, 16);
    if (kt + 4 < KT_OUT)                          // stream Wd beyond the pipeline
      __builtin_prefetch(bBase + (long long)(kt + 4) * NT_OUT * 512, 0, 1);
    #pragma unroll
    for (int m = 0; m < 4; ++m)                   // consume buf0
      acc[m] = wmma_bf16(A0[m], Bb0, acc[m]);
    if (kt + 2 < KT_OUT) {
      load_frag(Bb0, bBase + (long long)(kt + 2) * NT_OUT * 512, 8);  // issue buf0
      #pragma unroll
      for (int m = 0; m < 4; ++m)
        load_frag(A0[m], aBase + m * 16 * U_ + (kt + 2) * 32, 16);
    }
    #pragma unroll
    for (int m = 0; m < 4; ++m)                   // consume buf1
      acc[m] = wmma_bf16(A1[m], Bb1, acc[m]);
  }

  #pragma unroll
  for (int m = 0; m < 4; ++m)
    #pragma unroll
    for (int r = 0; r < 8; ++r) {
      const int brow = m * 16 + r + ((ln >= 16) ? 8 : 0);
      out[((long long)brow * T_ + t) * V_ + ncol] = acc[m][r];
    }
}

// ---------------- host orchestration ----------------
extern "C" void kernel_launch(void* const* d_in, const int* in_sizes, int n_in,
                              void* d_out, int out_size, void* d_ws, size_t ws_size,
                              hipStream_t stream) {
  const int*   tokens = (const int*)  d_in[0];   // [64,32]
  const float* enc    = (const float*)d_in[1];   // [64,64,1024]
  const float* h0     = (const float*)d_in[2];   // [64,1024]
  const float* c0     = (const float*)d_in[3];   // [64,1024]
  const float* emb    = (const float*)d_in[4];   // [32000,512]
  const float* Wk     = (const float*)d_in[5];   // [1536,4096]
  const float* Wr     = (const float*)d_in[6];   // [1024,4096]
  const float* bl     = (const float*)d_in[7];   // [4096]
  const float* Wd     = (const float*)d_in[8];   // [1024,32000]
  const float* bd     = (const float*)d_in[9];   // [32000]
  float* out = (float*)d_out;                    // [64,32,32000]

  char* ws = (char*)d_ws;
  size_t off = 0;
  auto carve = [&](size_t bytes) {
    size_t o = off;
    off = (off + bytes + 255) & ~(size_t)255;
    return o;
  };
  unsigned short* Wkr_sw  = (unsigned short*)(ws + carve((size_t)K_LSTM * N_LSTM * 2)); // 21.0 MB
  unsigned short* Wd_sw   = (unsigned short*)(ws + carve((size_t)U_ * V_ * 2));         // 65.5 MB
  float*          h_state = (float*)(ws + carve((size_t)B_ * U_ * 4));
  float*          c_state = (float*)(ws + carve((size_t)B_ * U_ * 4));
  unsigned short* h_bf    = (unsigned short*)(ws + carve((size_t)B_ * U_ * 2));
  unsigned short* xcat    = (unsigned short*)(ws + carve((size_t)B_ * K_LSTM * 2));
  (void)ws_size; (void)in_sizes; (void)n_in; (void)out_size;

  // one-time (per launch) weight conversion: bf16, pre-swizzled B fragments
  k_swizzle_wkr<<<(K_LSTM * N_LSTM) / 256, 256, 0, stream>>>(Wk, Wr, Wkr_sw);
  k_swizzle_wd <<<(U_ * V_) / 256,       256, 0, stream>>>(Wd, Wd_sw);
  k_init       <<<(B_ * U_) / 256,       256, 0, stream>>>(h0, c0, h_state, c_state);

  for (int t = 0; t < T_; ++t) {
    k_attn<<<B_,       256, 0, stream>>>(enc, h_state, tokens, emb, xcat, t);
    k_lstm<<<64,       256, 0, stream>>>(xcat, Wkr_sw, bl, h_state, c_state, h_bf);
    k_out <<<NT_OUT/8, 256, 0, stream>>>(h_bf, Wd_sw, bd, out, t);
  }
}
